// base_approximator_77171972374563
// MI455X (gfx1250) — matrix operand order repaired and checked by reading, not compiled, not run on hardware
//
#include <hip/hip_runtime.h>
#include <hip/hip_bf16.h>

// ---------------------------------------------------------------------------
// B=32, N=4096, C=4, K=256, D=2
// out = x - Phi (Phi^T Phi + aI)^-1 Phi^T x,  Phi[b,n,k] = exp(-w_k^2 |g-mu_k|^2)
//
// Dominant cost: P = Phi^T Phi (17.2 GFLOP) -> v_wmma_f32_16x16x32_f16.
// Numerics: P is severely ill-conditioned (alpha=1e-5 regularizer), so raw
// fp16 Phi (~5e-4 rel err) is not acceptable. Split-f16 (Markidis):
//   Phi = H + L  (both fp16),  P ~= H^T H + H^T L + L^T H   (f32 accum)
// gives ~2^-21 input accuracy while keeping all heavy math on the fp16 WMMA
// pipe (3x fp16 cost << 1x f32-WMMA cost). P is symmetric: compute only the
// 10 upper 64x64 tiles and mirror-store. Both Phi planes (2x67MB) fit in L2.
// ---------------------------------------------------------------------------

#define BB   32
#define NN   4096
#define CC   4
#define KK   256
#define ALPHA 1e-5f
#define LOG2E 1.4426950408889634f

typedef __attribute__((ext_vector_type(16))) _Float16     v16h;
typedef __attribute__((ext_vector_type(8)))  float        v8f;
typedef __attribute__((ext_vector_type(4))) unsigned int  u32x4;
typedef __attribute__((ext_vector_type(4)))  float        f32x4;

// ------------------------- Phi construction --------------------------------
// PhiH/PhiL[b][k][n]: split-f16 planes of Phi^T (n contiguous).
__global__ __launch_bounds__(256) void phi_kernel(
    const float* __restrict__ grid, const float* __restrict__ mu,
    const float* __restrict__ w, _Float16* __restrict__ PhiH,
    _Float16* __restrict__ PhiL) {
  __shared__ float sMuX[KK], sMuY[KK], sW2[KK];
  const int t = threadIdx.x;
  sMuX[t] = mu[2 * t + 0];
  sMuY[t] = mu[2 * t + 1];
  { float wv = w[t]; sW2[t] = wv * wv; }
  __syncthreads();

  const int gid = blockIdx.x * 256 + t;      // 0 .. B*N-1 (512 blocks)
  const int b = gid >> 12;
  const int n = gid & (NN - 1);
  const float gx = grid[(size_t)gid * 2 + 0];
  const float gy = grid[(size_t)gid * 2 + 1];
  _Float16* dH = PhiH + ((size_t)b * KK) * NN + n;
  _Float16* dL = PhiL + ((size_t)b * KK) * NN + n;
#pragma unroll 4
  for (int k = 0; k < KK; ++k) {
    const float dx = gx - sMuX[k];
    const float dy = gy - sMuY[k];
    const float d2 = dx * dx + dy * dy;
    const float val = exp2f(-LOG2E * sW2[k] * d2);   // v_exp_f32
    const _Float16 h = (_Float16)val;
    const _Float16 l = (_Float16)(val - (float)h);   // residual plane
    dH[(size_t)k * NN] = h;
    dL[(size_t)k * NN] = l;
  }
}

// ------------------- P = Phi^T Phi, split-f16 WMMA -------------------------
#define NCH 64  // n-chunk staged in LDS per iteration (4 slabs -> 32 KB)

__device__ inline v16h ld_frag16x2(const _Float16* p) {
  struct Pair { u32x4 a, b; } t;
  t.a = *(const u32x4*)(p);
  t.b = *(const u32x4*)(p + 8);
  return __builtin_bit_cast(v16h, t);
}
__device__ inline v16h ld_frag8_8(const _Float16* p0, const _Float16* p1) {
  struct Pair { u32x4 a, b; } t;
  t.a = *(const u32x4*)(p0);
  t.b = *(const u32x4*)(p1);
  return __builtin_bit_cast(v16h, t);
}
__device__ inline v8f wmma16(const v16h a, const v16h b, v8f c) {
  return __builtin_amdgcn_wmma_f32_16x16x32_f16(
      false, a, false, b, (short)0, c, false, false);
}

__global__ __launch_bounds__(256) void pgemm_kernel(
    const _Float16* __restrict__ PhiH, const _Float16* __restrict__ PhiL,
    float* __restrict__ P) {
  __shared__ u32x4 AsH4[64 * NCH / 8];
  __shared__ u32x4 AsL4[64 * NCH / 8];
  __shared__ u32x4 BsH4[64 * NCH / 8];
  __shared__ u32x4 BsL4[64 * NCH / 8];
  _Float16* AsH = (_Float16*)AsH4;
  _Float16* AsL = (_Float16*)AsL4;
  _Float16* BsH = (_Float16*)BsH4;
  _Float16* BsL = (_Float16*)BsL4;

  // 10 upper-triangular 64x64 tiles of the symmetric 256x256 P
  static const int TI[10] = {0, 0, 0, 0, 1, 1, 1, 2, 2, 3};
  static const int TJ[10] = {0, 1, 2, 3, 1, 2, 3, 2, 3, 3};
  const int b     = blockIdx.y;
  const int kBase = TI[blockIdx.x] * 64;
  const int lBase = TJ[blockIdx.x] * 64;
  const bool diag = (kBase == lBase);

  const int t    = threadIdx.x;
  const int lane = t & 31;
  const int wave = t >> 5;               // 8 waves
  const int wrow = wave & 3;             // subtile row (k dir)
  const int wcp  = wave >> 2;            // col pair: cols {2*wcp, 2*wcp+1}
  const int m    = lane & 15;
  const int hi   = lane >> 4;

  const _Float16* baseAH = PhiH + ((size_t)(b * KK + kBase)) * NN;
  const _Float16* baseAL = PhiL + ((size_t)(b * KK + kBase)) * NN;
  const _Float16* baseBH = PhiH + ((size_t)(b * KK + lBase)) * NN;
  const _Float16* baseBL = PhiL + ((size_t)(b * KK + lBase)) * NN;

  v8f acc0 = {};
  v8f acc1 = {};

  for (int n0 = 0; n0 < NN; n0 += NCH) {
    // stage 64 rows x 64 halves for all 4 planes (b128 loads -> LDS)
    for (int i = t; i < 64 * NCH / 8; i += 256) {
      const int row = i >> 3;            // NCH/8 = 8 uint4 per row
      const int cu  = i & 7;
      const size_t g = (size_t)row * NN + n0 + cu * 8;
      AsH4[i] = *(const u32x4*)(baseAH + g);
      AsL4[i] = *(const u32x4*)(baseAL + g);
      BsH4[i] = *(const u32x4*)(baseBH + g);
      BsL4[i] = *(const u32x4*)(baseBL + g);
    }
    __syncthreads();

#pragma unroll
    for (int s = 0; s < NCH; s += 32) {
      // A fragments (16x32 f16): lane m = row; 16 contiguous K per lane half
      const int aoff = (wrow * 16 + m) * NCH + s + hi * 16;
      const v16h aH = ld_frag16x2(AsH + aoff);
      const v16h aL = ld_frag16x2(AsL + aoff);
      // B fragments (32x16 f16): lane m = col; 16 contiguous K per lane half
      const int b0 = ((2 * wcp + 0) * 16 + m) * NCH + s + hi * 16;
      const int b1 = ((2 * wcp + 1) * 16 + m) * NCH + s + hi * 16;
      const v16h bH0 = ld_frag16x2(BsH + b0);
      const v16h bH1 = ld_frag16x2(BsH + b1);
      const v16h bL0 = ld_frag16x2(BsL + b0);
      const v16h bL1 = ld_frag16x2(BsL + b1);
      // P ~= H^T H + H^T L + L^T H  (drop L^T L, ~2^-22)
      acc0 = wmma16(aH, bH0, acc0);
      acc0 = wmma16(aH, bL0, acc0);
      acc0 = wmma16(aL, bH0, acc0);
      acc1 = wmma16(aH, bH1, acc1);
      acc1 = wmma16(aH, bL1, acc1);
      acc1 = wmma16(aL, bH1, acc1);
    }
    __syncthreads();
  }

  // C/D layout: VGPR r -> M = (lane>=16 ? 8 : 0) + r, N = lane & 15
  const int nC = lane & 15;
  const int mB = hi * 8;
  float* Pb = P + (size_t)b * KK * KK;
#pragma unroll
  for (int r = 0; r < 8; ++r) {
    const int row  = kBase + wrow * 16 + mB + r;
    const int col0 = lBase + (2 * wcp + 0) * 16 + nC;
    const int col1 = lBase + (2 * wcp + 1) * 16 + nC;
    Pb[(size_t)row * KK + col0] = acc0[r];
    Pb[(size_t)row * KK + col1] = acc1[r];
    if (!diag) {                          // mirror into the lower triangle
      Pb[(size_t)col0 * KK + row] = acc0[r];
      Pb[(size_t)col1 * KK + row] = acc1[r];
    }
  }
}

// ------------------------- coeff = Phi^T x ---------------------------------
__global__ __launch_bounds__(256) void coeff_kernel(
    const _Float16* __restrict__ PhiH, const _Float16* __restrict__ PhiL,
    const float* __restrict__ x, float* __restrict__ coeff) {
  const int t = threadIdx.x, lane = t & 31, wave = t >> 5;
  const int wid = blockIdx.x * 8 + wave;   // one wave per (b,k)
  const int b = wid >> 8;
  const int k = wid & (KK - 1);
  const _Float16* pH = PhiH + ((size_t)(b * KK + k)) * NN;
  const _Float16* pL = PhiL + ((size_t)(b * KK + k)) * NN;
  const f32x4* xv = (const f32x4*)(x + (size_t)b * NN * CC);
  float a0 = 0.f, a1 = 0.f, a2 = 0.f, a3 = 0.f;
  for (int n = lane; n < NN; n += 32) {
    const float p = (float)pH[n] + (float)pL[n];   // fp32-accurate Phi
    const f32x4 v = xv[n];
    a0 += p * v.x; a1 += p * v.y; a2 += p * v.z; a3 += p * v.w;
  }
#pragma unroll
  for (int off = 16; off > 0; off >>= 1) {
    a0 += __shfl_down(a0, off, 32);
    a1 += __shfl_down(a1, off, 32);
    a2 += __shfl_down(a2, off, 32);
    a3 += __shfl_down(a3, off, 32);
  }
  if (lane == 0) {
    f32x4 r; r.x = a0; r.y = a1; r.z = a2; r.w = a3;
    ((f32x4*)coeff)[(size_t)b * KK + k] = r;
  }
}

// --------------- Gauss-Jordan inverse of (P + aI), then c = Pinv*coeff -----
// One 256-thread workgroup per batch; 256x257 f32 matrix in 263 KB dynamic
// LDS (CDNA5 WGP: 320 KB). Stride 257 => conflict-free row ownership and
// same-address broadcast pivot-row reads. P SPD (+alpha) => no pivoting.
__global__ __launch_bounds__(256) void inv_solve_kernel(
    const float* __restrict__ P, const float* __restrict__ coeff,
    float* __restrict__ cOut) {
  extern __shared__ float M[];           // [256][257]
  const int LDM = KK + 1;
  const int b = blockIdx.x;
  const int r = threadIdx.x;

  const float* Pb = P + (size_t)b * KK * KK;
  for (int idx = r; idx < KK * KK; idx += 256) {
    const int row = idx >> 8, col = idx & (KK - 1);
    M[row * LDM + col] = Pb[idx] + (row == col ? ALPHA : 0.f);
  }
  __syncthreads();

  for (int p = 0; p < KK; ++p) {
    const float pivinv = 1.0f / M[p * LDM + p];
    const float f = M[r * LDM + p];
    __syncthreads();
    if (r == p) {
#pragma unroll 8
      for (int j = 0; j < KK; ++j) M[p * LDM + j] *= pivinv;
      M[p * LDM + p] = pivinv;
    }
    __syncthreads();
    if (r != p) {
#pragma unroll 8
      for (int j = 0; j < KK; ++j) M[r * LDM + j] -= f * M[p * LDM + j];
      M[r * LDM + p] = -f * pivinv;
    }
    __syncthreads();
  }

  const f32x4* co = (const f32x4*)coeff + (size_t)b * KK;
  float a0 = 0.f, a1 = 0.f, a2 = 0.f, a3 = 0.f;
#pragma unroll 4
  for (int l = 0; l < KK; ++l) {
    const float mv = M[r * LDM + l];
    const f32x4 v = co[l];
    a0 += mv * v.x; a1 += mv * v.y; a2 += mv * v.z; a3 += mv * v.w;
  }
  f32x4 out; out.x = a0; out.y = a1; out.z = a2; out.w = a3;
  ((f32x4*)cOut)[(size_t)b * KK + r] = out;
}

// ------------------------- out = x - Phi*c ---------------------------------
__global__ __launch_bounds__(256) void proj_kernel(
    const _Float16* __restrict__ PhiH, const _Float16* __restrict__ PhiL,
    const float* __restrict__ x, const float* __restrict__ cIn,
    float* __restrict__ out) {
  __shared__ float sc[KK * CC];
  const int blk = blockIdx.x;            // 512 = 32 batches * 16 chunks
  const int b = blk >> 4;
  const int nBase = (blk & 15) << 8;
  const int t = threadIdx.x;
  ((f32x4*)sc)[t] = ((const f32x4*)cIn)[(size_t)b * KK + t];
  __syncthreads();

  const int n = nBase + t;
  const _Float16* pH = PhiH + (size_t)b * KK * NN + n;
  const _Float16* pL = PhiL + (size_t)b * KK * NN + n;
  float a0 = 0.f, a1 = 0.f, a2 = 0.f, a3 = 0.f;
#pragma unroll 8
  for (int k = 0; k < KK; ++k) {
    const float p = (float)pH[(size_t)k * NN] + (float)pL[(size_t)k * NN];
    a0 += p * sc[k * 4 + 0];
    a1 += p * sc[k * 4 + 1];
    a2 += p * sc[k * 4 + 2];
    a3 += p * sc[k * 4 + 3];
  }
  const size_t oi = (size_t)b * NN + n;
  const f32x4 xv = ((const f32x4*)x)[oi];
  f32x4 o; o.x = xv.x - a0; o.y = xv.y - a1; o.z = xv.z - a2; o.w = xv.w - a3;
  ((f32x4*)out)[oi] = o;
}

// ---------------------------------------------------------------------------
extern "C" void kernel_launch(void* const* d_in, const int* in_sizes, int n_in,
                              void* d_out, int out_size, void* d_ws, size_t ws_size,
                              hipStream_t stream) {
  const float* x    = (const float*)d_in[0];  // [B,N,C]
  const float* grid = (const float*)d_in[1];  // [B,N,2]
  const float* mu   = (const float*)d_in[2];  // [K,2]
  const float* w    = (const float*)d_in[3];  // [K,1]

  char* ws = (char*)d_ws;
  const size_t phiBytes   = (size_t)BB * KK * NN * sizeof(_Float16); // 67.1 MB
  const size_t pBytes     = (size_t)BB * KK * KK * sizeof(float);    //  8.4 MB
  const size_t coeffBytes = (size_t)BB * KK * CC * sizeof(float);    //  131 KB
  _Float16* PhiH  = (_Float16*)ws;
  _Float16* PhiL  = (_Float16*)(ws + phiBytes);
  float*    P     = (float*)(ws + 2 * phiBytes);
  float*    coeff = (float*)(ws + 2 * phiBytes + pBytes);
  float*    cbuf  = (float*)(ws + 2 * phiBytes + pBytes + coeffBytes);

  phi_kernel  <<<(BB * NN) / 256, 256, 0, stream>>>(grid, mu, w, PhiH, PhiL);
  pgemm_kernel<<<dim3(10, BB), 256, 0, stream>>>(PhiH, PhiL, P);
  coeff_kernel<<<(BB * KK) / 8, 256, 0, stream>>>(PhiH, PhiL, x, coeff);
  inv_solve_kernel<<<BB, 256, (KK * (KK + 1)) * sizeof(float), stream>>>(P, coeff, cbuf);
  proj_kernel <<<(BB * NN) / 256, 256, 0, stream>>>(PhiH, PhiL, x, cbuf, (float*)d_out);
}
